// KDPointTrans_ablation_78898549228175
// MI455X (gfx1250) — matrix operand order repaired and checked by reading, not compiled, not run on hardware
//
#include <hip/hip_runtime.h>
#include <hip/hip_bf16.h>

typedef __attribute__((ext_vector_type(16))) _Float16 v16h;
typedef __attribute__((ext_vector_type(8)))  float    v8f;
typedef __attribute__((ext_vector_type(4)))  unsigned int uint32x4;
typedef __attribute__((ext_vector_type(8)))  int      int32x8;
typedef __attribute__((ext_vector_type(4)))  int      int32x4;

#define BB 8
#define NN 131072
#define CC 3
#define PRE 2048
#define POST 512
#define IOU_TH 0.7f
#define SCORE_TH 0.1f
#define NEGV -1000000000.0f
#define PAD_KEY -3.0e38f

#if defined(__has_builtin)
#if __has_builtin(__builtin_amdgcn_tensor_load_to_lds) && __has_builtin(__builtin_amdgcn_s_wait_tensorcnt)
#define USE_TDM 1
#endif
#endif

#ifdef USE_TDM
// Build a D# for a contiguous 1-D row copy (nelts fp32) and issue the TDM load.
// Descriptor layout per CDNA5 ISA:
//  g0: [1:0]count=1, [63:32]lds_addr, [120:64]global_addr, [127:126]type=2
//  g1: [17:16]data_size=2 (4B), [79:48]tensor_dim0, [111:80]tensor_dim1=1,
//      [127:112]tile_dim0, [143:128]tile_dim1=1, [207:160]tensor_dim0_stride
__device__ __forceinline__ void tdm_load_row(const float* gsrc, void* ldst, int nelts)
{
    unsigned lds_off = (unsigned)(size_t)ldst;          // generic->LDS offset (low 32b)
    unsigned long long ga = (unsigned long long)(size_t)gsrc;
    uint32x4 g0;
    g0[0] = 1u;                                          // count=1, user descriptor
    g0[1] = lds_off;                                     // lds_addr
    g0[2] = (unsigned)(ga & 0xffffffffu);                // global_addr[31:0]
    g0[3] = (unsigned)((ga >> 32) & 0x1ffffffu) | (2u << 30); // ga[56:32] | type=2
    int32x8 g1;
    g1[0] = (2 << 16);                                   // data_size=4B, wg_mask=0
    g1[1] = (nelts & 0xffff) << 16;                      // tensor_dim0[15:0]
    g1[2] = ((nelts >> 16) & 0xffff) | (1 << 16);        // tensor_dim0[31:16] | tensor_dim1=1 (lo)
    g1[3] = (nelts & 0xffff) << 16;                      // tensor_dim1 hi=0 | tile_dim0
    g1[4] = 1;                                           // tile_dim1=1, tile_dim2=0
    g1[5] = nelts;                                       // tensor_dim0_stride lo
    g1[6] = 0;                                           // stride hi, dim1_stride lo
    g1[7] = 0;
    int32x4 gz = {0, 0, 0, 0};
#if __clang_major__ >= 23
    int32x8 z8 = {0, 0, 0, 0, 0, 0, 0, 0};
    __builtin_amdgcn_tensor_load_to_lds(g0, g1, gz, gz, z8, 0);
#else
    __builtin_amdgcn_tensor_load_to_lds(g0, g1, gz, gz, 0);
#endif
}
#endif

// ---------------------------------------------------------------------------
// K1: per-batch top-PRE selection (sorted descending) via LDS bitonic merge.
// ---------------------------------------------------------------------------
__global__ __launch_bounds__(1024) void topk_kernel(
    const float* __restrict__ cls_tea, const float* __restrict__ box_tea,
    int* __restrict__ top_i,
    float* __restrict__ px1, float* __restrict__ px2,
    float* __restrict__ py1, float* __restrict__ py2,
    float* __restrict__ par, float* __restrict__ pval)
{
    __shared__ float key[2 * PRE];
    __shared__ int   kid[2 * PRE];
    const int b   = blockIdx.x;
    const int tid = threadIdx.x;

    for (int l = tid; l < PRE; l += 1024) { key[l] = PAD_KEY; kid[l] = 0; }
    __syncthreads();

    const float* cbp = cls_tea + (size_t)b * NN * CC;
    for (int T = 0; T < NN / PRE; ++T) {
        for (int l = tid; l < PRE; l += 1024) {
            int gi = T * PRE + l;
            const float* cp = cbp + (size_t)gi * CC;
            float m = fmaxf(cp[0], fmaxf(cp[1], cp[2]));
            key[PRE + l] = (m >= SCORE_TH) ? m : NEGV;
            kid[PRE + l] = gi;
            if (gi + PRE < NN)
                __builtin_prefetch(cbp + (size_t)(gi + PRE) * CC, 0, 0);
        }
        __syncthreads();
        for (int k = 2; k <= PRE; k <<= 1) {
            for (int j = k >> 1; j > 0; j >>= 1) {
                for (int l = tid; l < PRE; l += 1024) {
                    int p = l ^ j;
                    if (p > l) {
                        int A = PRE + l, Bq = PRE + p;
                        float ka = key[A], kb = key[Bq];
                        int   ia = kid[A], ib = kid[Bq];
                        bool aAfterB = (ka < kb) || (ka == kb && ia > ib);
                        bool desc = ((l & k) == 0);
                        bool sw = desc ? aAfterB : !aAfterB;
                        if (sw) { key[A] = kb; key[Bq] = ka; kid[A] = ib; kid[Bq] = ia; }
                    }
                }
                __syncthreads();
            }
        }
        for (int l = tid; l < PRE; l += 1024) {
            int q = 2 * PRE - 1 - l;
            float ka = key[l], kb = key[q];
            int   ia = kid[l], ib = kid[q];
            bool aAfterB = (ka < kb) || (ka == kb && ia > ib);
            if (aAfterB) { key[l] = kb; key[q] = ka; kid[l] = ib; kid[q] = ia; }
        }
        __syncthreads();
        for (int j = PRE >> 1; j > 0; j >>= 1) {
            for (int l = tid; l < PRE; l += 1024) {
                int p = l ^ j;
                if (p > l) {
                    float ka = key[l], kb = key[p];
                    int   ia = kid[l], ib = kid[p];
                    bool aAfterB = (ka < kb) || (ka == kb && ia > ib);
                    if (aAfterB) { key[l] = kb; key[p] = ka; kid[l] = ib; kid[p] = ia; }
                }
            }
            __syncthreads();
        }
    }

    const float* bbp = box_tea + (size_t)b * NN * 7;
    for (int l = tid; l < PRE; l += 1024) {
        int gi = kid[l];
        const float* bp = bbp + (size_t)gi * 7;
        float x = bp[0], y = bp[1], dx = bp[3], dy = bp[4];
        int o = b * PRE + l;
        px1[o] = x - 0.5f * dx;  px2[o] = x + 0.5f * dx;
        py1[o] = y - 0.5f * dy;  py2[o] = y + 0.5f * dy;
        par[o] = dx * dy;
        pval[o] = (key[l] > 0.5f * NEGV) ? 1.0f : 0.0f;
        top_i[o] = gi;
    }
}

// ---------------------------------------------------------------------------
// K2: suppression indicator matrix S^T[b][n][k] = (iou(k,n)>0.7 && k<n), fp16.
// ---------------------------------------------------------------------------
__global__ __launch_bounds__(256) void pairs_kernel(
    const float* __restrict__ px1, const float* __restrict__ px2,
    const float* __restrict__ py1, const float* __restrict__ py2,
    const float* __restrict__ par, _Float16* __restrict__ S)
{
    size_t idx = (size_t)blockIdx.x * 256 + threadIdx.x;
    int k = (int)(idx & (PRE - 1));
    int n = (int)((idx >> 11) & (PRE - 1));
    int b = (int)(idx >> 22);
    int ok = b * PRE + k, on = b * PRE + n;
    float iw = fminf(px2[ok], px2[on]) - fmaxf(px1[ok], px1[on]);
    float ih = fminf(py2[ok], py2[on]) - fmaxf(py1[ok], py1[on]);
    iw = fmaxf(iw, 0.0f); ih = fmaxf(ih, 0.0f);
    float inter = iw * ih;
    float denom = fmaxf(par[ok] + par[on] - inter, 1e-6f);
    bool sup = (inter > IOU_TH * denom) && (k < n);
    S[(size_t)(b * PRE + n) * PRE + k] = sup ? (_Float16)1.0f : (_Float16)0.0f;
}

// ---------------------------------------------------------------------------
// K3: greedy NMS, one full wave32 per batch; WMMA matvec pulls suppression
// from all previously kept boxes; LDS primed via TDM (tensor_load_to_lds).
// ---------------------------------------------------------------------------
__global__ __launch_bounds__(32) void nms_kernel(
    const float* __restrict__ px1, const float* __restrict__ px2,
    const float* __restrict__ py1, const float* __restrict__ py2,
    const float* __restrict__ par, const float* __restrict__ pval,
    const _Float16* __restrict__ S, const int* __restrict__ top_i,
    int* __restrict__ sel, float* __restrict__ msk)
{
    __shared__ float sx1[PRE], sx2[PRE], sy1[PRE], sy2[PRE], sar[PRE], sva[PRE];
    __shared__ _Float16 skeep[PRE];
    const int b    = blockIdx.x;
    const int L    = threadIdx.x;
    const int half = L >> 4;
    const int col  = L & 15;
    const int bo   = b * PRE;

#ifdef USE_TDM
    // Tensor Data Mover: six 8KB 1-D row copies global->LDS, then wait on
    // TENSORcnt. Issued once per wave (EXEC ignored by TDM ops).
    tdm_load_row(px1 + bo, sx1, PRE);
    tdm_load_row(px2 + bo, sx2, PRE);
    tdm_load_row(py1 + bo, sy1, PRE);
    tdm_load_row(py2 + bo, sy2, PRE);
    tdm_load_row(par + bo, sar, PRE);
    tdm_load_row(pval + bo, sva, PRE);
    for (int i = L; i < PRE; i += 32) skeep[i] = (_Float16)0.0f;
    __builtin_amdgcn_s_wait_tensorcnt(0);
    __syncthreads();
#else
    for (int i = L; i < PRE; i += 32) {
        sx1[i] = px1[bo + i]; sx2[i] = px2[bo + i];
        sy1[i] = py1[bo + i]; sy2[i] = py2[bo + i];
        sar[i] = par[bo + i]; sva[i] = pval[bo + i];
        skeep[i] = (_Float16)0.0f;
    }
    __syncthreads();
#endif

    const _Float16* Sb = S + (size_t)bo * PRE;
    for (int t = 0; t < PRE / 16; ++t) {
        const int base  = t * 16;
        const int nglob = base + col;
        v8f c = {0.f, 0.f, 0.f, 0.f, 0.f, 0.f, 0.f, 0.f};
        const int nch = (base + 31) >> 5;        // undecided keep flags are 0
        const _Float16* Srow = Sb + (size_t)nglob * PRE;
        for (int ch = 0; ch < nch; ++ch) {
            const int cb = ch * 32;
            v16h a;
            #pragma unroll
            for (int e = 0; e < 8; ++e) a[e] = skeep[cb + 16 * half + e];
            #pragma unroll
            for (int e = 0; e < 8; ++e) a[8 + e] = skeep[cb + 16 * half + 8 + e];
            v16h bm = *(const v16h*)(Srow + cb + 16 * half);
            c = __builtin_amdgcn_wmma_f32_16x16x32_f16(
                    false, a, false, bm, (short)0, c, false, false);
        }
        float cnt = c[0];

        float x1 = sx1[nglob], x2 = sx2[nglob];
        float y1 = sy1[nglob], y2 = sy2[nglob];
        float ar = sar[nglob], va = sva[nglob];
        float supf = 0.0f;
        for (int i = 0; i < 16; ++i) {
            float mycand = (va > 0.5f && cnt < 0.5f && supf < 0.5f) ? 1.0f : 0.0f;
            float ki = __shfl(mycand, i, 32);
            if (ki > 0.5f) {
                int gi = base + i;
                float iw = fminf(x2, sx2[gi]) - fmaxf(x1, sx1[gi]);
                float ih = fminf(y2, sy2[gi]) - fmaxf(y1, sy1[gi]);
                iw = fmaxf(iw, 0.0f); ih = fmaxf(ih, 0.0f);
                float inter = iw * ih;
                float denom = fmaxf(ar + sar[gi] - inter, 1e-6f);
                if (inter > IOU_TH * denom && col > i) supf = 1.0f;
            }
        }
        float kept = (va > 0.5f && cnt < 0.5f && supf < 0.5f) ? 1.0f : 0.0f;
        if (half == 0) skeep[nglob] = (_Float16)kept;
        __syncthreads();
    }

    if (L == 0) {
        int pos = 0;
        for (int i = 0; i < PRE; ++i) {
            if ((float)skeep[i] > 0.5f && pos < POST) {
                sel[b * POST + pos] = top_i[bo + i];
                msk[b * POST + pos] = 1.0f;
                ++pos;
            }
        }
        for (; pos < POST; ++pos) { sel[b * POST + pos] = 0; msk[b * POST + pos] = 0.0f; }
    }
}

// ---------------------------------------------------------------------------
// K4: gathers + all ten outputs (flat-concatenated fp32, 98304 elements).
// ---------------------------------------------------------------------------
__global__ __launch_bounds__(256) void gather_kernel(
    const float* __restrict__ box_tea, const float* __restrict__ cls_tea,
    const float* __restrict__ box_stu, const float* __restrict__ cls_stu,
    const float* __restrict__ cls_preds, const float* __restrict__ rcnn_tea,
    const int* __restrict__ sel, const float* __restrict__ msk,
    float* __restrict__ out)
{
    int j = blockIdx.x * 256 + threadIdx.x;
    if (j >= BB * POST) return;
    int b = j / POST;
    int s = sel[j];
    float m = msk[j];
    size_t gb = (size_t)b * NN + (size_t)s;

    float* o_kd_stu   = out;           // (B*POST,1)
    float* o_kd_tea   = out + 4096;    // (B*POST,1)
    float* o_rois_tea = out + 8192;    // (B,POST,7)
    float* o_sc_tea   = out + 36864;   // (B,POST)
    float* o_lb_tea   = out + 40960;   // (B,POST)
    float* o_rois_stu = out + 45056;   // (B,POST,7)
    float* o_sc_stu   = out + 73728;   // (B,POST)
    float* o_lb_stu   = out + 77824;   // (B,POST)
    float* o_cls_sel  = out + 81920;   // (B,POST,3)
    float* o_selmask  = out + 94208;   // (B*POST,1)

    const float* bt = box_tea + gb * 7;
    const float* bs = box_stu + gb * 7;
    #pragma unroll
    for (int c = 0; c < 7; ++c) {
        o_rois_tea[j * 7 + c] = bt[c] * m;
        o_rois_stu[j * 7 + c] = bs[c] * m;
    }
    const float* ct = cls_tea + gb * 3;
    float t0 = ct[0], t1 = ct[1], t2 = ct[2];
    float mt = t0; int lt = 0;
    if (t1 > mt) { mt = t1; lt = 1; }
    if (t2 > mt) { mt = t2; lt = 2; }
    o_sc_tea[j] = mt * m;
    o_lb_tea[j] = (m > 0.0f) ? (float)(lt + 1) : 1.0f;

    const float* cs = cls_stu + gb * 3;
    float u0 = cs[0], u1 = cs[1], u2 = cs[2];
    float mu = u0; int lu = 0;
    if (u1 > mu) { mu = u1; lu = 1; }
    if (u2 > mu) { mu = u2; lu = 2; }
    o_sc_stu[j] = mu * m;
    o_lb_stu[j] = (m > 0.0f) ? (float)(lu + 1) : 1.0f;

    const float* cp = cls_preds + gb * 3;
    float c0 = cp[0] * m, c1 = cp[1] * m, c2 = cp[2] * m;
    o_cls_sel[j * 3 + 0] = c0;
    o_cls_sel[j * 3 + 1] = c1;
    o_cls_sel[j * 3 + 2] = c2;
    o_kd_stu[j] = fmaxf(c0, fmaxf(c1, c2)) * m;
    o_kd_tea[j] = rcnn_tea[j] * m;
    o_selmask[j] = m;
}

extern "C" void kernel_launch(void* const* d_in, const int* in_sizes, int n_in,
                              void* d_out, int out_size, void* d_ws, size_t ws_size,
                              hipStream_t stream)
{
    const float* box_tea   = (const float*)d_in[0];
    const float* cls_tea   = (const float*)d_in[1];
    const float* box_stu   = (const float*)d_in[2];
    const float* cls_stu   = (const float*)d_in[3];
    const float* cls_preds = (const float*)d_in[4];
    const float* rcnn_tea  = (const float*)d_in[5];

    char* ws = (char*)d_ws;
    int*      top_i = (int*)  (ws + 0 * (1 << 16));
    float*    px1   = (float*)(ws + 1 * (1 << 16));
    float*    px2   = (float*)(ws + 2 * (1 << 16));
    float*    py1   = (float*)(ws + 3 * (1 << 16));
    float*    py2   = (float*)(ws + 4 * (1 << 16));
    float*    par   = (float*)(ws + 5 * (1 << 16));
    float*    pva   = (float*)(ws + 6 * (1 << 16));
    int*      sel   = (int*)  (ws + 7 * (1 << 16));
    float*    msk   = (float*)(ws + 7 * (1 << 16) + (1 << 14));
    _Float16* S     = (_Float16*)(ws + (1 << 19));

    topk_kernel<<<BB, 1024, 0, stream>>>(cls_tea, box_tea, top_i,
                                         px1, px2, py1, py2, par, pva);
    pairs_kernel<<<(BB * PRE * PRE) / 256, 256, 0, stream>>>(px1, px2, py1, py2, par, S);
    nms_kernel<<<BB, 32, 0, stream>>>(px1, px2, py1, py2, par, pva, S, top_i, sel, msk);
    gather_kernel<<<(BB * POST + 255) / 256, 256, 0, stream>>>(
        box_tea, cls_tea, box_stu, cls_stu, cls_preds, rcnn_tea, sel, msk, (float*)d_out);
}
